// LiquidS4Cell_88149908783123
// MI455X (gfx1250) — matrix operand order repaired and checked by reading, not compiled, not run on hardware
//
#include <hip/hip_runtime.h>

// ---------------------------------------------------------------------------
// LiquidS4 cell for MI455X (gfx1250, wave32).
//   Phase 1: chunked parallel scan of the diagonal recurrence
//            x_t = a*x_{t-1} + (dt*B)*u_t      (exact f32 math)
//   Phase 2: y = xs @ Q via V_WMMA_F32_16X16X32_F16 with split-f16 operands
//            (hi/lo decomposition: 3 WMMAs ~= f32 accuracy at f16 speed).
// Workspace layout (needs ~65.5 MB):
//   xs_hi  : 32768*512 f16   (32 MB)
//   xs_lo  : 32768*512 f16   (32 MB)
//   qt_hi  : 512*512  f16 transposed [d][n]  (512 KB)
//   qt_lo  : 512*512  f16 transposed [d][n]  (512 KB)
//   cf, cs : 8*16*512 f32 chunk finals / chunk starts (256 KB each)
// ---------------------------------------------------------------------------

typedef _Float16 v16h __attribute__((ext_vector_type(16)));
typedef _Float16 v8h  __attribute__((ext_vector_type(8)));
typedef float    v8f  __attribute__((ext_vector_type(8)));

#define D_MODEL 512
#define N_STATE 512
#define BATCH   8
#define SEQ     4096
#define M_TOTAL (BATCH * SEQ)      // 32768 rows of the GEMM
#define NCHUNK  16
#define CLEN    (SEQ / NCHUNK)     // 256

// ---------------- Phase 1a: split + transpose Q ----------------------------
__global__ void qsplit_kernel(const float* __restrict__ Q,
                              _Float16* __restrict__ qth,
                              _Float16* __restrict__ qtl) {
    int t = blockIdx.x * blockDim.x + threadIdx.x;   // 512*512 threads
    int n = t >> 9;          // row of Q  (contraction index)
    int d = t & 511;         // col of Q  (output index) -> coalesced read
    float q = Q[n * D_MODEL + d];
    _Float16 h = (_Float16)q;
    _Float16 l = (_Float16)(q - (float)h);
    qth[d * N_STATE + n] = h;      // store transposed [d][n]
    qtl[d * N_STATE + n] = l;
}

// ---------------- Phase 1b: per-chunk finals (zero init) -------------------
__global__ void chunk_final_kernel(const float* __restrict__ u,
                                   const float* __restrict__ Lambda,
                                   const float* __restrict__ Bvec,
                                   const float* __restrict__ log_dt,
                                   float* __restrict__ cf) {
    int t = blockIdx.x * blockDim.x + threadIdx.x;   // BATCH*NCHUNK*512
    int n = t & 511;
    int c = (t >> 9) & (NCHUNK - 1);
    int b = t >> 13;
    float dt  = expf(log_dt[0]);
    float a   = expf(-dt * expf(Lambda[n]));
    float bdt = dt * Bvec[n];
    const float* up = u + ((size_t)(b * SEQ + c * CLEN)) * N_STATE + n;
    float x = 0.0f;
#pragma unroll 8
    for (int i = 0; i < CLEN; ++i)
        x = fmaf(x, a, bdt * up[(size_t)i * N_STATE]);
    cf[(b * NCHUNK + c) * N_STATE + n] = x;
}

// ---------------- Phase 1c: scan chunk finals -> chunk start states --------
__global__ void chunk_scan_kernel(const float* __restrict__ Lambda,
                                  const float* __restrict__ log_dt,
                                  const float* __restrict__ cf,
                                  float* __restrict__ cs) {
    int t = blockIdx.x * blockDim.x + threadIdx.x;   // BATCH*512
    int n = t & 511;
    int b = t >> 9;
    float dt = expf(log_dt[0]);
    float aC = expf(-dt * expf(Lambda[n]) * (float)CLEN);  // a^CLEN exactly
    float s = 0.0f;
#pragma unroll
    for (int c = 0; c < NCHUNK; ++c) {
        int idx = (b * NCHUNK + c) * N_STATE + n;
        cs[idx] = s;                      // state entering chunk c
        s = fmaf(s, aC, cf[idx]);
    }
}

// ---------------- Phase 1d: expand chunks, emit split-f16 xs ---------------
__global__ void chunk_expand_kernel(const float* __restrict__ u,
                                    const float* __restrict__ Lambda,
                                    const float* __restrict__ Bvec,
                                    const float* __restrict__ log_dt,
                                    const float* __restrict__ cs,
                                    _Float16* __restrict__ xsh,
                                    _Float16* __restrict__ xsl) {
    int t = blockIdx.x * blockDim.x + threadIdx.x;   // BATCH*NCHUNK*512
    int n = t & 511;
    int c = (t >> 9) & (NCHUNK - 1);
    int b = t >> 13;
    float dt  = expf(log_dt[0]);
    float a   = expf(-dt * expf(Lambda[n]));
    float bdt = dt * Bvec[n];
    float x = cs[(b * NCHUNK + c) * N_STATE + n];
    size_t base = ((size_t)(b * SEQ + c * CLEN)) * N_STATE + n;
#pragma unroll 4
    for (int i = 0; i < CLEN; ++i) {
        size_t idx = base + (size_t)i * N_STATE;
        x = fmaf(x, a, bdt * u[idx]);
        _Float16 h = (_Float16)x;
        xsh[idx] = h;
        xsl[idx] = (_Float16)(x - (float)h);
    }
}

// ---------------- Phase 2: y = xs @ Q via WMMA f16 split -------------------
#define BM 128
#define BN 128
#define BK 32
#define LDA 40   // padded LDS row stride (f16) -> conflict-free b128 frags
#define LDB 40

union V16 { v8h h[2]; v16h v; };

__global__ __launch_bounds__(256)
void s4_gemm_kernel(const _Float16* __restrict__ xsh,
                    const _Float16* __restrict__ xsl,
                    const _Float16* __restrict__ qth,
                    const _Float16* __restrict__ qtl,
                    float* __restrict__ out) {
    __shared__ __align__(16) _Float16 sah[BM * LDA];
    __shared__ __align__(16) _Float16 sal[BM * LDA];
    __shared__ __align__(16) _Float16 sbh[BN * LDB];
    __shared__ __align__(16) _Float16 sbl[BN * LDB];

    const int tid  = threadIdx.x;
    const int lane = tid & 31;
    const int w    = tid >> 5;     // 8 waves
    const int wm   = w >> 1;       // 0..3  -> M offset wm*32
    const int wn   = w & 1;        // 0..1  -> N offset wn*64
    const int lr   = lane & 15;
    const int lh   = lane >> 4;

    const int m0 = blockIdx.x * BM;
    const int n0 = blockIdx.y * BN;

    v8f acc[2][4] = {};

    for (int kt = 0; kt < N_STATE / BK; ++kt) {
        __syncthreads();
        // ---- stage tiles: straight 16B copies (Q pre-transposed to [d][n])
        {
            int q0 = tid, q1 = tid + 256;           // 512 quads per array
            int r0 = q0 >> 2, g0 = q0 & 3;
            int r1 = q1 >> 2, g1 = q1 & 3;
            size_t ga0 = (size_t)(m0 + r0) * N_STATE + kt * BK + g0 * 8;
            size_t ga1 = (size_t)(m0 + r1) * N_STATE + kt * BK + g1 * 8;
            *(uint4*)&sah[r0 * LDA + g0 * 8] = *(const uint4*)&xsh[ga0];
            *(uint4*)&sah[r1 * LDA + g1 * 8] = *(const uint4*)&xsh[ga1];
            *(uint4*)&sal[r0 * LDA + g0 * 8] = *(const uint4*)&xsl[ga0];
            *(uint4*)&sal[r1 * LDA + g1 * 8] = *(const uint4*)&xsl[ga1];
            size_t gb0 = (size_t)(n0 + r0) * N_STATE + kt * BK + g0 * 8;
            size_t gb1 = (size_t)(n0 + r1) * N_STATE + kt * BK + g1 * 8;
            *(uint4*)&sbh[r0 * LDB + g0 * 8] = *(const uint4*)&qth[gb0];
            *(uint4*)&sbh[r1 * LDB + g1 * 8] = *(const uint4*)&qth[gb1];
            *(uint4*)&sbl[r0 * LDB + g0 * 8] = *(const uint4*)&qtl[gb0];
            *(uint4*)&sbl[r1 * LDB + g1 * 8] = *(const uint4*)&qtl[gb1];
        }
        __syncthreads();

        // ---- fragments per the ISA 16-bit A(16x32) / B(32x16) layouts
        v16h ah[2], al[2], bh[4], bl[4];
#pragma unroll
        for (int i = 0; i < 2; ++i) {
            int row = wm * 32 + i * 16 + lr;        // A: M = lane&15
            int ab  = lh * 8;                       // K base: lanes>=16 -> +8
            const _Float16* pah = &sah[row * LDA];
            const _Float16* pal = &sal[row * LDA];
            V16 th, tl;
            th.h[0] = *(const v8h*)(pah + ab);       // elems 0..7  = K base..base+7
            th.h[1] = *(const v8h*)(pah + 16 + ab);  // elems 8..15 = K 16+base..
            tl.h[0] = *(const v8h*)(pal + ab);
            tl.h[1] = *(const v8h*)(pal + 16 + ab);
            ah[i] = th.v; al[i] = tl.v;
        }
#pragma unroll
        for (int j = 0; j < 4; ++j) {
            int col = wn * 64 + j * 16 + lr;        // B: N = lane&15
            const _Float16* pbh = &sbh[col * LDB + lh * 8];   // lanes>=16: K base +8
            const _Float16* pbl = &sbl[col * LDB + lh * 8];
            V16 th, tl;
            th.h[0] = *(const v8h*)(pbh);
            th.h[1] = *(const v8h*)(pbh + 16);
            tl.h[0] = *(const v8h*)(pbl);
            tl.h[1] = *(const v8h*)(pbl + 16);
            bh[j] = th.v; bl[j] = tl.v;
        }

        // ---- 3-product split-f16 MACs, f32 accumulation
#pragma unroll
        for (int i = 0; i < 2; ++i)
#pragma unroll
            for (int j = 0; j < 4; ++j) {
                acc[i][j] = __builtin_amdgcn_wmma_f32_16x16x32_f16(
                    false, al[i], false, bh[j], (short)0, acc[i][j], false, false);
                acc[i][j] = __builtin_amdgcn_wmma_f32_16x16x32_f16(
                    false, ah[i], false, bl[j], (short)0, acc[i][j], false, false);
                acc[i][j] = __builtin_amdgcn_wmma_f32_16x16x32_f16(
                    false, ah[i], false, bh[j], (short)0, acc[i][j], false, false);
            }
    }

    // ---- epilogue: C/D layout VGPR r -> M=r (lanes 0-15) / M=8+r (16-31)
#pragma unroll
    for (int i = 0; i < 2; ++i)
#pragma unroll
        for (int j = 0; j < 4; ++j) {
            int col     = n0 + wn * 64 + j * 16 + lr;
            int rowbase = m0 + wm * 32 + i * 16 + lh * 8;
#pragma unroll
            for (int r = 0; r < 8; ++r)
                out[(size_t)(rowbase + r) * D_MODEL + col] = acc[i][j][r];
        }
}

// ---------------------------------------------------------------------------
extern "C" void kernel_launch(void* const* d_in, const int* in_sizes, int n_in,
                              void* d_out, int out_size, void* d_ws, size_t ws_size,
                              hipStream_t stream) {
    const float* u      = (const float*)d_in[0];
    const float* Lambda = (const float*)d_in[1];
    const float* Bvec   = (const float*)d_in[2];
    const float* Q      = (const float*)d_in[3];
    const float* log_dt = (const float*)d_in[4];
    float* out = (float*)d_out;

    char* ws = (char*)d_ws;
    const size_t XS_BYTES = (size_t)M_TOTAL * N_STATE * 2;   // 32 MB each
    _Float16* xsh = (_Float16*)(ws);
    _Float16* xsl = (_Float16*)(ws + XS_BYTES);
    _Float16* qth = (_Float16*)(ws + 2 * XS_BYTES);
    _Float16* qtl = (_Float16*)(ws + 2 * XS_BYTES + (size_t)N_STATE * D_MODEL * 2);
    float*    cf  = (float*)   (ws + 2 * XS_BYTES + (size_t)N_STATE * D_MODEL * 4);
    float*    cs  = cf + BATCH * NCHUNK * N_STATE;
    (void)in_sizes; (void)n_in; (void)out_size; (void)ws_size;

    qsplit_kernel<<<(N_STATE * D_MODEL) / 256, 256, 0, stream>>>(Q, qth, qtl);
    chunk_final_kernel<<<(BATCH * NCHUNK * N_STATE) / 256, 256, 0, stream>>>(
        u, Lambda, Bvec, log_dt, cf);
    chunk_scan_kernel<<<(BATCH * N_STATE) / 256, 256, 0, stream>>>(
        Lambda, log_dt, cf, cs);
    chunk_expand_kernel<<<(BATCH * NCHUNK * N_STATE) / 256, 256, 0, stream>>>(
        u, Lambda, Bvec, log_dt, cs, xsh, xsl);
    dim3 grid(M_TOTAL / BM, D_MODEL / BN);
    s4_gemm_kernel<<<grid, 256, 0, stream>>>(xsh, xsl, qth, qtl, out);
}
